// StructNet_75118978007047
// MI455X (gfx1250) — matrix operand order repaired and checked by reading, not compiled, not run on hardware
//
#include <hip/hip_runtime.h>
#include <cstdint>
#include <cstddef>

// ---------------------------------------------------------------------------
// Types for CDNA5 WMMA
// ---------------------------------------------------------------------------
typedef __attribute__((ext_vector_type(16))) __bf16 v16bf;
typedef __attribute__((ext_vector_type(8)))  float  v8f;

#define TILE_E 16      // edges per block tile (one wave32 per block)

// ---------------------------------------------------------------------------
// CDNA5 async global->LDS helpers (ASYNCcnt path, CDNA5 ISA)
// ---------------------------------------------------------------------------
__device__ __forceinline__ void async_g2l_b128(const void* g, void* l) {
  unsigned loff = (unsigned)(size_t)l;               // low 32 bits = LDS offset
  unsigned long long ga = (unsigned long long)g;
  asm volatile("global_load_async_to_lds_b128 %0, %1, off"
               :: "v"(loff), "v"(ga) : "memory");
}
__device__ __forceinline__ void async_g2l_b32(const void* g, void* l) {
  unsigned loff = (unsigned)(size_t)l;
  unsigned long long ga = (unsigned long long)g;
  asm volatile("global_load_async_to_lds_b32 %0, %1, off"
               :: "v"(loff), "v"(ga) : "memory");
}
__device__ __forceinline__ void wait_async0() {
  asm volatile("s_wait_asynccnt 0x0" ::: "memory");
}

// ---------------------------------------------------------------------------
// Argument structs
// ---------------------------------------------------------------------------
struct AttEdgeArgs {
  const float* r;            // [E]
  const float* w;            // [E,4]
  const int*   src;          // [E]
  const float* feats[4];     // input feats per di position t
  const float* basis[4][4];  // [di][do]
  int ndi;
  int dis[4];
  int w1o[2][4][4];          // [kv][t][do] element offsets into params (f32)
  int b1o[2][4][4];
  int b2o[2][4][4];
  int w2p[2][4][4];          // element offsets into packed bf16 W2 buffer
  float* k_out;              // [E,4,16]
  float* v_out;              // [E,4,16]
  long E;
};

struct ConvEdgeArgs {
  const float* r;
  const float* w;
  const int*   src;
  const int*   dst;
  const float* feats[4];     // per di
  const float* basis[4];     // (di, do=0)
  int w1o[4], b1o[4], b2o[4];
  int w2p[4];
  float* agg;                // [N,64]
  long E;
};

struct QArgs {
  const float* feats[4];
  int ndi; int dis[4];
  int wqo[4];
  float* q;                  // [N,4,16]
  long N;
};

struct LogitArgs {
  const float* q;            // [N,4,16]
  const float* k;            // [E,4,16]
  const int* dst;
  float* logit;              // [E]
  unsigned* segmax;          // [N]
  float inv;
  int ndi; int dis[4];
  long E;
};

struct ProjArgs {
  const float* featsIn[4];
  int present[4];
  int wproj[4];
  int wskip[4];
  float* featsOut[4];
  const float* agg;          // [N,4,16]
  long N;
};

struct NormArgs {
  float* f[4];
  int wo[4], bo[4];
  long N;
};

// ---------------------------------------------------------------------------
// Utility kernels
// ---------------------------------------------------------------------------
__global__ void k_zero32(unsigned* __restrict__ p, long n) {
  long i = (long)blockIdx.x * blockDim.x + threadIdx.x;
  if (i < n) p[i] = 0u;
}

// Pack one radial W2 (f32, [32, outc] row-major) into WMMA B-fragment order:
// packed[nt*512 + lane*16 + e] = W2[(hi*8+e+(e&8))*outc + nt*16 + (lane&15)]
__global__ void k_packW2(const float* __restrict__ src, __bf16* __restrict__ dst,
                         int outc) {
  int idx = blockIdx.x * blockDim.x + threadIdx.x;
  int total = outc * 32;
  if (idx >= total) return;
  int e    = idx & 15;
  int lane = (idx >> 4) & 31;
  int nt   = idx >> 9;
  int r16 = lane & 15, hi = lane >> 4;
  dst[idx] = (__bf16)src[(hi * 8 + e + (e & 8)) * outc + nt * 16 + r16];
}

// ---------------------------------------------------------------------------
// Fused attention edge-conv. Per 16-edge tile, per (do, di) pair:
//   - stage basis tile to LDS once via async global->LDS (reused by k and v)
//   - per kv: hidden = relu(x@W1+b1) computed per-lane directly in A-fragment
//     order (no LDS round trip), R = hidden @ W2 via bf16 WMMA in two column
//     halves (half h2 maps exactly to output channels {2*h2, 2*h2+1})
//   - contraction acc[o,p] += sum_{i,f,m} R[o,i,f]*bs[p,m,f]*fe[i,m]
// ---------------------------------------------------------------------------
__global__ __launch_bounds__(32)
void k_att_edge(AttEdgeArgs A, const float* __restrict__ pf,
                const __bf16* __restrict__ pw) {
  __shared__ float xe[TILE_E][5];
  __shared__ int   esrc[TILE_E];
  __shared__ float fe[TILE_E][16][16];            // [e][i][m-slot]   16 KB
  __shared__ float Rl[TILE_E][224];               // half R tile      14 KB
  __shared__ alignas(16) float bsl[TILE_E * 343]; // basis tile       22 KB

  const int lane = threadIdx.x;
  const int r16  = lane & 15;
  const int hi   = lane >> 4;
  const long e0  = (long)blockIdx.x * TILE_E;

  if (lane < TILE_E) {
    long e = e0 + lane;
    bool ok = e < A.E;
    esrc[lane] = ok ? A.src[e] : 0;
    xe[lane][0] = ok ? A.r[e] : 0.f;
    for (int j = 0; j < 4; ++j) xe[lane][1 + j] = ok ? A.w[e * 4 + j] : 0.f;
  }
  __syncthreads();

  // Gather source-node features for all input degrees once per tile.
  for (int t = 0; t < A.ndi; ++t) {
    int di = A.dis[t];
    int mdim = 2 * di + 1, moff = di * di;
    const float* f = A.feats[t];
    int tot = TILE_E * 16 * mdim;
    for (int idx = lane; idx < tot; idx += 32) {
      int e = idx / (16 * mdim);
      int rem = idx - e * 16 * mdim;
      int i = rem / mdim;
      int m = rem - i * mdim;
      fe[e][i][moff + m] = f[((long)esrc[e] * 16 + i) * mdim + m];
    }
  }
  __syncthreads();

  const bool eOK = (e0 + r16) < A.E;
  int valid_e = (int)((A.E - e0) < TILE_E ? (A.E - e0) : TILE_E);

  for (int dd = 0; dd < 4; ++dd) {
    const int pdim = 2 * dd + 1;
    float acc[2][2][7];                          // [kv][h2][p]
    for (int a = 0; a < 2; ++a)
      for (int b = 0; b < 2; ++b)
        for (int p = 0; p < 7; ++p) acc[a][b][p] = 0.f;

    for (int t = 0; t < A.ndi; ++t) {
      const int di = A.dis[t];
      const int mn = di < dd ? di : dd;
      const int nf = 2 * mn + 1;
      const int mdim = 2 * di + 1, moff = di * di;
      const int pmn = pdim * mdim * nf;

      // ---- stage basis tile into LDS (async global->LDS) ----
      {
        const char* gb = (const char*)A.basis[di][dd] + (size_t)e0 * pmn * 4;
        char* lb = (char*)bsl;
        int vbytes = valid_e * pmn * 4;
        int n128 = vbytes >> 4;
        for (int c = lane; c < n128; c += 32)
          async_g2l_b128(gb + (size_t)c * 16, lb + (size_t)c * 16);
        int tailw = (vbytes & 15) >> 2;
        if (lane < tailw)
          async_g2l_b32(gb + (size_t)n128 * 16 + lane * 4,
                        lb + (size_t)n128 * 16 + lane * 4);
        wait_async0();
      }
      __syncthreads();

      for (int kv = 0; kv < 2; ++kv) {
        const float* W1 = pf + A.w1o[kv][t][dd];
        const float* b1 = pf + A.b1o[kv][t][dd];
        const float* b2 = pf + A.b2o[kv][t][dd];
        const v16bf* Wp = (const v16bf*)(pw + A.w2p[kv][t][dd]);

        // hidden directly in A-fragment order (row r16, K = hi*8+e+(e&8))
        v16bf afrag;
#pragma unroll
        for (int e = 0; e < 16; ++e) {
          int kk = hi * 8 + e + (e & 8);
          float h = b1[kk];
#pragma unroll
          for (int j = 0; j < 5; ++j) h += xe[r16][j] * W1[j * 32 + kk];
          afrag[e] = (__bf16)(h > 0.f ? h : 0.f);
        }

        for (int h2 = 0; h2 < 2; ++h2) {
          const int tb = h2 * 2 * nf;            // first global col-tile
          for (int nt2 = 0; nt2 < 2 * nf; ++nt2) {
            __builtin_prefetch(Wp + (size_t)(tb + nt2 + 1) * 32 + lane, 0, 3);
            v16bf bfrag = Wp[(size_t)(tb + nt2) * 32 + lane];
            v8f cfr = {};
            cfr = __builtin_amdgcn_wmma_f32_16x16x32_bf16(
                false, afrag, false, bfrag, (short)0, cfr, false, false);
            float badd = b2[(tb + nt2) * 16 + r16];
#pragma unroll
            for (int v2 = 0; v2 < 8; ++v2)
              Rl[v2 + hi * 8][nt2 * 16 + r16] = cfr[v2] + badd;
          }
          __syncthreads();

          // contraction: lane handles edge r16, output channel o = h2*2 + hi
          if (eOK) {
            const float* bse = bsl + r16 * pmn;
            for (int f2 = 0; f2 < nf; ++f2) {
              for (int m = 0; m < mdim; ++m) {
                float tsum = 0.f;
#pragma unroll
                for (int i = 0; i < 16; ++i)
                  tsum += Rl[r16][(hi * 16 + i) * nf + f2] * fe[r16][i][moff + m];
                for (int p = 0; p < pdim; ++p)
                  acc[kv][h2][p] += tsum * bse[(p * mdim + m) * nf + f2];
              }
            }
          }
          __syncthreads();                       // Rl reused next half / kv / t
        }
      }
    }

    if (eOK) {
      int moffo = dd * dd;
      long ebase = e0 + r16;
      for (int kv = 0; kv < 2; ++kv) {
        float* outp = kv ? A.v_out : A.k_out;
        for (int h2 = 0; h2 < 2; ++h2) {
          int o = h2 * 2 + hi;
          for (int p = 0; p < pdim; ++p)
            outp[(ebase * 4 + o) * 16 + moffo + p] = acc[kv][h2][p];
        }
      }
    }
  }
}

// ---------------------------------------------------------------------------
// Fused conv edge kernel (out degree 0, cout=64):
//   g[i]   = sum_m bs[e,0,m,0]*fe[i,m]
//   out[o] = sum_i (hidden @ W2 + b2)[o*16+i] * g[i], accumulated over di
// ---------------------------------------------------------------------------
__global__ __launch_bounds__(32)
void k_conv_edge(ConvEdgeArgs A, const float* __restrict__ pf,
                 const __bf16* __restrict__ pw) {
  __shared__ float xe[TILE_E][5];
  __shared__ int   esrc[TILE_E];
  __shared__ int   edst[TILE_E];
  __shared__ float gall[4][TILE_E][16];
  __shared__ float Rl[TILE_E][256];

  const int lane = threadIdx.x;
  const int r16 = lane & 15, hi = lane >> 4;
  const long e0 = (long)blockIdx.x * TILE_E;

  if (lane < TILE_E) {
    long e = e0 + lane;
    bool ok = e < A.E;
    esrc[lane] = ok ? A.src[e] : 0;
    edst[lane] = ok ? A.dst[e] : 0;
    xe[lane][0] = ok ? A.r[e] : 0.f;
    for (int j = 0; j < 4; ++j) xe[lane][1 + j] = ok ? A.w[e * 4 + j] : 0.f;
  }
  __syncthreads();

  for (int di = 0; di < 4; ++di) {
    int mdim = 2 * di + 1;
    const float* f = A.feats[di];
    const float* bs = A.basis[di];
    for (int idx = lane; idx < TILE_E * 16; idx += 32) {
      int e = idx >> 4, i = idx & 15;
      float sAcc = 0.f;
      if (e0 + e < A.E) {
        long node = esrc[e];
        const float* bse = bs + (e0 + e) * mdim;
        for (int m = 0; m < mdim; ++m)
          sAcc += bse[m] * f[(node * 16 + i) * mdim + m];
      }
      gall[di][e][i] = sAcc;
    }
  }
  __syncthreads();

  const bool eOK = (e0 + r16) < A.E;

  for (int oc = 0; oc < 4; ++oc) {               // 16 output channels / chunk
    float acc[8];
#pragma unroll
    for (int z = 0; z < 8; ++z) acc[z] = 0.f;

    for (int di = 0; di < 4; ++di) {
      const float* W1 = pf + A.w1o[di];
      const float* b1 = pf + A.b1o[di];
      const float* b2 = pf + A.b2o[di];
      const v16bf* Wp = (const v16bf*)(pw + A.w2p[di]);

      v16bf afrag;
#pragma unroll
      for (int e = 0; e < 16; ++e) {
        int kk = hi * 8 + e + (e & 8);
        float h = b1[kk];
#pragma unroll
        for (int j = 0; j < 5; ++j) h += xe[r16][j] * W1[j * 32 + kk];
        afrag[e] = (__bf16)(h > 0.f ? h : 0.f);
      }

      for (int nt = 0; nt < 16; ++nt) {
        int gt = oc * 16 + nt;                   // global column tile
        v16bf bfrag = Wp[(size_t)gt * 32 + lane];
        v8f cfr = {};
        cfr = __builtin_amdgcn_wmma_f32_16x16x32_bf16(
            false, afrag, false, bfrag, (short)0, cfr, false, false);
        float badd = b2[gt * 16 + r16];
#pragma unroll
        for (int v2 = 0; v2 < 8; ++v2)
          Rl[v2 + hi * 8][nt * 16 + r16] = cfr[v2] + badd;
      }
      __syncthreads();

      for (int oo = 0; oo < 8; ++oo) {
        int ol = hi * 8 + oo;
        float s2 = 0.f;
#pragma unroll
        for (int i = 0; i < 16; ++i)
          s2 += Rl[r16][ol * 16 + i] * gall[di][r16][i];
        acc[oo] += s2;
      }
      __syncthreads();
    }

    if (eOK) {
      long node = edst[r16];
      for (int oo = 0; oo < 8; ++oo)
        atomicAdd(&A.agg[node * 64 + oc * 16 + hi * 8 + oo], acc[oo]);
    }
  }
}

// ---------------------------------------------------------------------------
// Node-level kernels
// ---------------------------------------------------------------------------
__global__ void k_qnode(QArgs A, const float* __restrict__ pf) {
  long idx = (long)blockIdx.x * blockDim.x + threadIdx.x;
  if (idx >= A.N * 4) return;
  long n = idx >> 2;
  int d = (int)(idx & 3);
  for (int t = 0; t < A.ndi; ++t) {
    int di = A.dis[t], mdim = 2 * di + 1, moff = di * di;
    const float* Wq = pf + A.wqo[t];
    const float* f = A.feats[t];
    for (int m = 0; m < mdim; ++m) {
      float s = 0.f;
      for (int c = 0; c < 16; ++c)
        s += f[(n * 16 + c) * mdim + m] * Wq[c * 4 + d];
      A.q[(n * 4 + d) * 16 + moff + m] = s;
    }
  }
}

__global__ void k_logit(LogitArgs A) {
  long e = (long)blockIdx.x * blockDim.x + threadIdx.x;
  if (e >= A.E) return;
  long nd = A.dst[e];
  float s = 0.f;
  for (int t = 0; t < A.ndi; ++t) {
    int di = A.dis[t], mdim = 2 * di + 1, moff = di * di;
    for (int d = 0; d < 4; ++d)
      for (int m = 0; m < mdim; ++m)
        s += A.q[(nd * 4 + d) * 16 + moff + m] *
             A.k[(e * 4 + d) * 16 + moff + m];
  }
  s *= A.inv;
  A.logit[e] = s;
  unsigned bits = __float_as_uint(s);
  unsigned enc = (bits & 0x80000000u) ? ~bits : (bits | 0x80000000u);
  atomicMax(A.segmax + nd, enc);
}

__global__ void k_alpha(const float* __restrict__ logit, const int* __restrict__ dst,
                        const unsigned* __restrict__ segmax,
                        float* __restrict__ alpha, float* __restrict__ segsum, long E) {
  long e = (long)blockIdx.x * blockDim.x + threadIdx.x;
  if (e >= E) return;
  long nd = dst[e];
  unsigned enc = segmax[nd];
  unsigned bits = (enc & 0x80000000u) ? (enc & 0x7FFFFFFFu) : ~enc;
  float m = __uint_as_float(bits);
  float a = __expf(logit[e] - m);
  alpha[e] = a;
  atomicAdd(segsum + nd, a);
}

__global__ void k_aggregate(const float* __restrict__ alpha,
                            const float* __restrict__ segsum,
                            const int* __restrict__ dst,
                            const float* __restrict__ v_edge,
                            float* __restrict__ agg, long E) {
  long idx = (long)blockIdx.x * blockDim.x + threadIdx.x;
  if (idx >= E * 64) return;
  long e = idx >> 6;
  int slot = (int)(idx & 63);
  long nd = dst[e];
  float a = alpha[e] / (segsum[nd] + 1e-9f);
  atomicAdd(&agg[nd * 64 + slot], a * v_edge[idx]);
}

__global__ void k_proj(ProjArgs A, const float* __restrict__ pf) {
  long idx = (long)blockIdx.x * blockDim.x + threadIdx.x;
  if (idx >= A.N * 64) return;
  long n = idx >> 6;
  int dk = (int)((idx >> 4) & 3);
  int c = (int)(idx & 15);
  int mdim = 2 * dk + 1, moff = dk * dk;
  const float* Wp = pf + A.wproj[dk];
  for (int m = 0; m < mdim; ++m) {
    float o = 0.f;
    for (int d = 0; d < 4; ++d)
      o += A.agg[(n * 4 + d) * 16 + moff + m] * Wp[d * 16 + c];
    if (A.present[dk]) {
      const float* Ws = pf + A.wskip[dk];
      const float* fi = A.featsIn[dk];
      for (int a2 = 0; a2 < 16; ++a2)
        o += fi[(n * 16 + a2) * mdim + m] * Ws[a2 * 16 + c];
    }
    A.featsOut[dk][(n * 16 + c) * mdim + m] = o;
  }
}

__global__ void k_norm(NormArgs A, const float* __restrict__ pf) {
  long idx = (long)blockIdx.x * blockDim.x + threadIdx.x;
  if (idx >= A.N * 4) return;
  long n = idx >> 2;
  int dk = (int)(idx & 3);
  int mdim = 2 * dk + 1;
  float* f = A.f[dk];
  const float* W = pf + A.wo[dk];
  const float* b = pf + A.bo[dk];
  float nrm[16];
  for (int c = 0; c < 16; ++c) {
    float s = 0.f;
    for (int m = 0; m < mdim; ++m) {
      float x = f[(n * 16 + c) * mdim + m];
      s += x * x;
    }
    nrm[c] = sqrtf(s + 1e-12f);
  }
  for (int c = 0; c < 16; ++c) {
    float sc = b[c];
    for (int cc = 0; cc < 16; ++cc) sc += nrm[cc] * W[cc * 16 + c];
    sc = sc > 0.f ? sc : 0.f;
    float mul = sc / nrm[c];
    for (int m = 0; m < mdim; ++m) f[(n * 16 + c) * mdim + m] *= mul;
  }
}

__global__ void k_selfbias(float* __restrict__ h, const float* __restrict__ f0,
                           const float* __restrict__ pf, int wself, int bself, long N) {
  long idx = (long)blockIdx.x * blockDim.x + threadIdx.x;
  if (idx >= N * 64) return;
  long n = idx >> 6;
  int o = (int)(idx & 63);
  float s = h[idx] + pf[bself + o];
  for (int c = 0; c < 16; ++c) s += f0[n * 16 + c] * pf[wself + c * 64 + o];
  h[idx] = s;
}

__global__ void k_pool(const float* __restrict__ h, const int* __restrict__ gid,
                       float* __restrict__ pooled, float scale, long N) {
  long idx = (long)blockIdx.x * blockDim.x + threadIdx.x;
  if (idx >= N * 64) return;
  long n = idx >> 6;
  int o = (int)(idx & 63);
  atomicAdd(&pooled[(long)gid[n] * 64 + o], h[idx] * scale);
}

__global__ void k_fc(const float* __restrict__ g, const float* __restrict__ pf,
                     int wo, int bo, float* __restrict__ out, int Bn) {
  int idx = blockIdx.x * blockDim.x + threadIdx.x;
  if (idx >= Bn * 64) return;
  int b = idx >> 6, j = idx & 63;
  float s = pf[bo + j];
  for (int i = 0; i < 64; ++i) s += g[b * 64 + i] * pf[wo + i * 64 + j];
  out[idx] = s > 0.f ? s : 0.f;
}

__global__ void k_dot(const float* __restrict__ ha, const float* __restrict__ hg,
                      float* __restrict__ out, int Bn) {
  int b = blockIdx.x * blockDim.x + threadIdx.x;
  if (b >= Bn) return;
  float s = 0.f;
  for (int c = 0; c < 64; ++c) s += ha[b * 64 + c] * hg[b * 64 + c];
  out[b] = s;
}

// ---------------------------------------------------------------------------
// Host-side parameter-offset walker (mirrors jax pytree sorted-key flatten)
// ---------------------------------------------------------------------------
struct LayerOff {
  int Wproj[4];
  int Wq[4];
  int Wskip[4];
  int rW1[2][4][4], rW2[2][4][4], rb1[2][4][4], rb2[2][4][4];  // [kv][t][do]
  int rW2p[2][4][4];                                           // packed offs
  int normW[4], normb[4];
  int ndi; int dis[4];
};
struct BranchOff {
  int conv_Wself, conv_bself;
  int cW1[4], cW2[4], cb1[4], cb2[4];
  int cW2p[4];
  int fc_W, fc_b;
  LayerOff L[2];
};

static long computeOffsets(BranchOff bo[2], long* packTotal) {
  long cur = 0, pcur = 0;
  auto take  = [&](long n) { long o = cur;  cur  += n; return (int)o; };
  auto ptake = [&](long n) { long o = pcur; pcur += n; return (int)o; };
  for (int br = 0; br < 2; ++br) {
    BranchOff& B = bo[br];
    // conv dict keys sorted: Wself, bself, r
    B.conv_Wself = take(16 * 64);
    B.conv_bself = take(64);
    for (int di = 0; di < 4; ++di) {  // r keys "0,0","1,0","2,0","3,0"
      B.cW1[di] = take(5 * 32);
      B.cW2[di] = take(32 * 1024);
      B.cW2p[di] = ptake(32 * 1024);
      B.cb1[di] = take(32);
      B.cb2[di] = take(1024);
    }
    B.fc_W = take(64 * 64);
    B.fc_b = take(64);
    for (int l = 0; l < 2; ++l) {
      LayerOff& L = B.L[l];
      L.ndi = (l == 0) ? 1 : 4;
      for (int t = 0; t < 4; ++t) L.dis[t] = (l == 0) ? 0 : t;
      // att keys sorted: Wproj, Wq, Wskip, rk, rv
      for (int d = 0; d < 4; ++d) L.Wproj[d] = take(4 * 16);
      for (int t = 0; t < L.ndi; ++t) L.Wq[t] = take(16 * 4);
      for (int t = 0; t < L.ndi; ++t) L.Wskip[t] = take(16 * 16);
      for (int kv = 0; kv < 2; ++kv) {
        for (int t = 0; t < L.ndi; ++t) {
          int di = L.dis[t];
          for (int dd = 0; dd < 4; ++dd) {  // keys "di,do" sorted di-major
            int mn = di < dd ? di : dd;
            int nf = 2 * mn + 1;
            L.rW1[kv][t][dd] = take(5 * 32);
            L.rW2[kv][t][dd] = take(32 * 64 * nf);
            L.rW2p[kv][t][dd] = ptake(32 * 64 * nf);
            L.rb1[kv][t][dd] = take(32);
            L.rb2[kv][t][dd] = take(64 * nf);
          }
        }
      }
      for (int d = 0; d < 4; ++d) L.normW[d] = take(16 * 16);
      for (int d = 0; d < 4; ++d) L.normb[d] = take(16);
    }
  }
  *packTotal = pcur;
  return cur;
}

// ---------------------------------------------------------------------------
// Launcher
// ---------------------------------------------------------------------------
static inline unsigned gblk(long n, int t) { return (unsigned)((n + t - 1) / t); }

extern "C" void kernel_launch(void* const* d_in, const int* in_sizes, int n_in,
                              void* d_out, int out_size, void* d_ws, size_t ws_size,
                              hipStream_t stream) {
  (void)n_in; (void)ws_size;
  const float* f_in_b[2]   = {(const float*)d_in[0], (const float*)d_in[4]};
  const float* r_b[2]      = {(const float*)d_in[1], (const float*)d_in[5]};
  const float* w_b[2]      = {(const float*)d_in[2], (const float*)d_in[6]};
  const float* basis_b[2]  = {(const float*)d_in[3], (const float*)d_in[7]};
  const float* params      = (const float*)d_in[8];
  const int*   src_b[2]    = {(const int*)d_in[9],  (const int*)d_in[11]};
  const int*   dst_b[2]    = {(const int*)d_in[10], (const int*)d_in[12]};
  const int*   graph_id    = (const int*)d_in[13];

  const long N = (long)in_sizes[0] / 16;   // [N,16,1]
  const long E = (long)in_sizes[1];        // [E,1]
  const int  Bn = out_size;                // [B,1]
  const long NPG = N / (Bn > 0 ? Bn : 1);

  BranchOff BO[2];
  long PACKTOT = 0;
  (void)computeOffsets(BO, &PACKTOT);

  // basis leaf offsets, sorted keys "di,do" di-major
  long basOff[4][4];
  {
    long c = 0;
    for (int di = 0; di < 4; ++di)
      for (int dd = 0; dd < 4; ++dd) {
        basOff[di][dd] = c;
        int mn = di < dd ? di : dd;
        c += E * (long)(2 * dd + 1) * (2 * di + 1) * (2 * mn + 1);
      }
  }

  // Workspace carve-up
  char* wp = (char*)d_ws;
  auto carve = [&](size_t bytes) {
    void* p = wp;
    wp += (bytes + 255) & ~(size_t)255;
    return p;
  };
  __bf16*   pwk     = (__bf16*)carve((size_t)PACKTOT * 2);
  float*    featsA  = (float*)carve((size_t)N * 256 * 4);
  float*    featsB  = (float*)carve((size_t)N * 256 * 4);
  float*    k_edge  = (float*)carve((size_t)E * 64 * 4);
  float*    v_edge  = (float*)carve((size_t)E * 64 * 4);
  float*    qbuf    = (float*)carve((size_t)N * 64 * 4);
  float*    logit   = (float*)carve((size_t)E * 4);
  float*    alphaE  = (float*)carve((size_t)E * 4);
  unsigned* segmax  = (unsigned*)carve((size_t)N * 4);
  float*    segsum  = (float*)carve((size_t)N * 4);
  float*    agg     = (float*)carve((size_t)N * 64 * 4);
  float*    convagg = (float*)carve((size_t)N * 64 * 4);
  float*    pooled  = (float*)carve((size_t)Bn * 64 * 4);
  float*    hfc[2]  = {(float*)carve((size_t)Bn * 64 * 4),
                       (float*)carve((size_t)Bn * 64 * 4)};

  auto zero32 = [&](void* p, long n) {
    k_zero32<<<gblk(n, 256), 256, 0, stream>>>((unsigned*)p, n);
  };
  auto pack = [&](int srcOff, int dstOff, int outc) {
    k_packW2<<<gblk((long)outc * 32, 256), 256, 0, stream>>>(
        params + srcOff, pwk + dstOff, outc);
  };

  // Pack all radial W2 matrices into WMMA B-fragment order (bf16)
  for (int br = 0; br < 2; ++br) {
    const BranchOff& B = BO[br];
    for (int di = 0; di < 4; ++di) pack(B.cW2[di], B.cW2p[di], 1024);
    for (int l = 0; l < 2; ++l) {
      const LayerOff& L = B.L[l];
      for (int kv = 0; kv < 2; ++kv)
        for (int t = 0; t < L.ndi; ++t) {
          int di = L.dis[t];
          for (int dd = 0; dd < 4; ++dd) {
            int mn = di < dd ? di : dd;
            pack(L.rW2[kv][t][dd], L.rW2p[kv][t][dd], 64 * (2 * mn + 1));
          }
        }
    }
  }

  const unsigned ETILES = gblk(E, TILE_E);

  for (int br = 0; br < 2; ++br) {
    const float* f_in = f_in_b[br];
    const float* rr = r_b[br];
    const float* ww = w_b[br];
    const float* bb = basis_b[br];
    const int* src = src_b[br];
    const int* dst = dst_b[br];
    const BranchOff& B = BO[br];

    float* featsA_d[4], *featsB_d[4];
    for (int d = 0; d < 4; ++d) {
      featsA_d[d] = featsA + (long)N * 16 * (d * d);
      featsB_d[d] = featsB + (long)N * 16 * (d * d);
    }

    for (int l = 0; l < 2; ++l) {
      const LayerOff& L = B.L[l];
      zero32(segmax, N);
      zero32(segsum, N);
      zero32(agg, N * 64);

      AttEdgeArgs aa = {};
      aa.r = rr; aa.w = ww; aa.src = src; aa.E = E;
      aa.ndi = L.ndi;
      for (int t = 0; t < 4; ++t) {
        aa.dis[t] = L.dis[t];
        aa.feats[t] = (l == 0) ? f_in : (const float*)featsA_d[t];
      }
      for (int di = 0; di < 4; ++di)
        for (int dd = 0; dd < 4; ++dd)
          aa.basis[di][dd] = bb + basOff[di][dd];
      for (int kv = 0; kv < 2; ++kv)
        for (int t = 0; t < 4; ++t)
          for (int dd = 0; dd < 4; ++dd) {
            aa.w1o[kv][t][dd] = L.rW1[kv][t][dd];
            aa.b1o[kv][t][dd] = L.rb1[kv][t][dd];
            aa.b2o[kv][t][dd] = L.rb2[kv][t][dd];
            aa.w2p[kv][t][dd] = L.rW2p[kv][t][dd];
          }
      aa.k_out = k_edge; aa.v_out = v_edge;
      k_att_edge<<<ETILES, 32, 0, stream>>>(aa, params, pwk);

      QArgs qa = {};
      qa.ndi = L.ndi; qa.N = N; qa.q = qbuf;
      for (int t = 0; t < 4; ++t) {
        qa.dis[t] = L.dis[t];
        qa.feats[t] = aa.feats[t];
        qa.wqo[t] = L.Wq[t];
      }
      k_qnode<<<gblk(N * 4, 256), 256, 0, stream>>>(qa, params);

      int d_tot = 0;
      for (int t = 0; t < L.ndi; ++t) d_tot += 4 * (2 * L.dis[t] + 1);
      LogitArgs la = {};
      la.q = qbuf; la.k = k_edge; la.dst = dst; la.logit = logit;
      la.segmax = segmax; la.inv = 1.0f / sqrtf((float)d_tot);
      la.ndi = L.ndi; la.E = E;
      for (int t = 0; t < 4; ++t) la.dis[t] = L.dis[t];
      k_logit<<<gblk(E, 256), 256, 0, stream>>>(la);

      k_alpha<<<gblk(E, 256), 256, 0, stream>>>(logit, dst, segmax, alphaE, segsum, E);
      k_aggregate<<<gblk(E * 64, 256), 256, 0, stream>>>(alphaE, segsum, dst,
                                                         v_edge, agg, E);

      ProjArgs pa = {};
      pa.agg = agg; pa.N = N;
      for (int dk = 0; dk < 4; ++dk) {
        pa.wproj[dk] = L.Wproj[dk];
        if (l == 0) {
          pa.present[dk] = (dk == 0) ? 1 : 0;
          pa.featsIn[dk] = (dk == 0) ? f_in : nullptr;
          pa.wskip[dk] = (dk == 0) ? L.Wskip[0] : 0;
        } else {
          pa.present[dk] = 1;
          pa.featsIn[dk] = featsA_d[dk];
          pa.wskip[dk] = L.Wskip[dk];
        }
        pa.featsOut[dk] = (l == 0) ? featsA_d[dk] : featsB_d[dk];
      }
      k_proj<<<gblk(N * 64, 256), 256, 0, stream>>>(pa, params);

      NormArgs na = {};
      na.N = N;
      for (int dk = 0; dk < 4; ++dk) {
        na.f[dk] = (l == 0) ? featsA_d[dk] : featsB_d[dk];
        na.wo[dk] = L.normW[dk];
        na.bo[dk] = L.normb[dk];
      }
      k_norm<<<gblk(N * 4, 256), 256, 0, stream>>>(na, params);
    }

    // ---- conv layer ----
    zero32(convagg, N * 64);
    ConvEdgeArgs ca = {};
    ca.r = rr; ca.w = ww; ca.src = src; ca.dst = dst; ca.E = E;
    for (int di = 0; di < 4; ++di) {
      ca.feats[di] = featsB_d[di];
      ca.basis[di] = bb + basOff[di][0];
      ca.w1o[di] = B.cW1[di]; ca.b1o[di] = B.cb1[di];
      ca.b2o[di] = B.cb2[di]; ca.w2p[di] = B.cW2p[di];
    }
    ca.agg = convagg;
    k_conv_edge<<<ETILES, 32, 0, stream>>>(ca, params, pwk);

    k_selfbias<<<gblk(N * 64, 256), 256, 0, stream>>>(convagg, featsB_d[0],
                                                      params, B.conv_Wself,
                                                      B.conv_bself, N);
    zero32(pooled, (long)Bn * 64);
    k_pool<<<gblk(N * 64, 256), 256, 0, stream>>>(convagg, graph_id, pooled,
                                                  1.0f / (float)NPG, N);
    k_fc<<<gblk((long)Bn * 64, 256), 256, 0, stream>>>(pooled, params, B.fc_W,
                                                       B.fc_b, hfc[br], Bn);
  }

  k_dot<<<gblk(Bn, 64), 64, 0, stream>>>(hfc[0], hfc[1], (float*)d_out, Bn);
}
